// InputToLeafSinkhorn_47966194761838
// MI455X (gfx1250) — matrix operand order repaired and checked by reading, not compiled, not run on hardware
//
#include <hip/hip_runtime.h>
#include <hip/hip_bf16.h>
#include <math.h>

// ---------------------------------------------------------------------------
// InputToLeafSinkhorn for MI455X (gfx1250, wave32, WMMA)
//
//   g  = -log(-log(u + eps) + eps)
//   A  = exp(logits + g)                         (T = 1)
//   sinkhorn == diagonal scaling:  r = 1/(A c),  c = 1/(A^T r),  c0 = 1
//   P  = diag(r) A diag(c)
//   out = x @ P^T        -> bf16 WMMA GEMM, f32 accumulate (memory-bound)
// ---------------------------------------------------------------------------

typedef __attribute__((ext_vector_type(16))) __bf16 v16bf;
typedef __attribute__((ext_vector_type(8)))  float  v8f;

#define BATCH   131072
#define NIN     512
#define NLV     512
#define ITERS   20
#define SBLK    16          // sinkhorn persistent blocks (must be co-resident)
#define SROWS   32          // rows of A per sinkhorn block
#define ASTRIDE 516         // padded row stride (floats) to avoid bank conflicts
#define GM      64          // GEMM rows per block

__device__ __forceinline__ unsigned short f2bf(float f) {
  unsigned int u = __float_as_uint(f);
  u += 0x7FFFu + ((u >> 16) & 1u);          // round-to-nearest-even
  return (unsigned short)(u >> 16);
}

// ---------------------------------------------------------------------------
// Init: zero per-iteration colsum buffers and barrier counters (graph-safe).
// ---------------------------------------------------------------------------
__global__ void init_ws_kernel(float* __restrict__ colsum, int* __restrict__ barct) {
  int t = blockIdx.x * blockDim.x + threadIdx.x;
  if (t < ITERS * NIN) colsum[t] = 0.0f;
  if (t < ITERS)       barct[t]  = 0;
}

// ---------------------------------------------------------------------------
// Persistent Sinkhorn: 16 blocks x 512 threads. Each block owns 32 rows of A
// in LDS. One device-wide barrier per iteration (c-vector exchange).
// Emits P as bf16 in the WMMA B-operand swizzle:
//   element(kt, nt, lane, e):  n = nt*16 + (lane&15),  k = kt*32 + (lane>>4)*16 + e
//   flat index = ((kt*32 + nt)*32 + lane)*16 + e
// ---------------------------------------------------------------------------
__global__ __launch_bounds__(512) void sinkhorn_kernel(
    const float* __restrict__ logits, const float* __restrict__ noise,
    unsigned short* __restrict__ Psw, float* __restrict__ colsum,
    int* __restrict__ barct)
{
  __shared__ __align__(16) float A[SROWS * ASTRIDE];   // 66 KB of 320 KB WGP LDS
  __shared__ float cvec[NIN];
  __shared__ float rvec[SROWS];
  __shared__ float red[512];

  const int t   = threadIdx.x;
  const int blk = blockIdx.x;
  const int li  = t >> 4;            // local row 0..31
  const int j0  = (t & 15) * 32;     // column base, 32 contiguous cols/thread

  // Build A = exp(logits + gumbel(noise)) for this block's 32 rows.
  const int gi = blk * SROWS + li;
  const float* lrow = logits + (size_t)gi * NIN + j0;
  const float* nrow = noise  + (size_t)gi * NIN + j0;
  #pragma unroll 8
  for (int s = 0; s < 32; ++s) {
    float u = nrow[s];
    float g = -__logf(-__logf(u + 1e-20f) + 1e-20f);
    A[li * ASTRIDE + j0 + s] = __expf(lrow[s] + g);
  }
  if (t < NIN) cvec[t] = 1.0f;
  __syncthreads();

  for (int it = 0; it < ITERS; ++it) {
    // ---- r_i = 1 / sum_j A[i][j] * c[j]  (local to this block's rows) ----
    float p = 0.0f;
    #pragma unroll 8
    for (int s = 0; s < 32; ++s) p += A[li * ASTRIDE + j0 + s] * cvec[j0 + s];
    red[t] = p;
    __syncthreads();
    if (t < SROWS) {
      float s = 0.0f;
      #pragma unroll
      for (int q = 0; q < 16; ++q) s += red[t * 16 + q];
      rvec[t] = 1.0f / s;
    }
    __syncthreads();

    // ---- partial column sums: sum_i r_i * A[i][j]  -> global atomics ----
    float cs = 0.0f;
    #pragma unroll 8
    for (int i = 0; i < SROWS; ++i) cs += rvec[i] * A[i * ASTRIDE + t];
    unsafeAtomicAdd(&colsum[it * NIN + t], cs);
    __threadfence();
    __syncthreads();

    // ---- device-wide barrier (per-iteration counter, pre-zeroed) ----
    if (t == 0) {
      __hip_atomic_fetch_add(&barct[it], 1, __ATOMIC_ACQ_REL, __HIP_MEMORY_SCOPE_AGENT);
      while (__hip_atomic_load(&barct[it], __ATOMIC_ACQUIRE, __HIP_MEMORY_SCOPE_AGENT) < SBLK) {
        __builtin_amdgcn_s_sleep(2);
      }
    }
    __syncthreads();

    // ---- c_j = 1 / colsum_j ----
    float tot = __hip_atomic_load(&colsum[it * NIN + t], __ATOMIC_RELAXED,
                                  __HIP_MEMORY_SCOPE_AGENT);
    cvec[t] = 1.0f / tot;
    __syncthreads();
  }

  // ---- write P = diag(r) A diag(c) as bf16 in WMMA-B swizzled layout ----
  {
    const int n  = blk * SROWS + li;       // leaf row of P
    const int kb = (t & 15) * 32;          // this thread's 32-wide K chunk
    const int kt = kb >> 5;
    const int nt = n >> 4;
    const int ll = n & 15;
    const float r = rvec[li];
    __align__(16) unsigned short tmp[32];
    #pragma unroll 8
    for (int kk = 0; kk < 32; ++kk)
      tmp[kk] = f2bf(r * A[li * ASTRIDE + kb + kk] * cvec[kb + kk]);
    unsigned short* d0 = Psw + (((size_t)(kt * 32 + nt) * 32) + ll) * 16;       // K 0..15
    unsigned short* d1 = Psw + (((size_t)(kt * 32 + nt) * 32) + 16 + ll) * 16;  // K 16..31
    ((uint4*)d0)[0] = *(const uint4*)&tmp[0];
    ((uint4*)d0)[1] = *(const uint4*)&tmp[8];
    ((uint4*)d1)[0] = *(const uint4*)&tmp[16];
    ((uint4*)d1)[1] = *(const uint4*)&tmp[24];
  }
}

// ---------------------------------------------------------------------------
// GEMM: out[B,512] = x[B,512] @ P^T, bf16 WMMA, f32 accumulate.
// Block: 256 threads = 8 waves; tile M=64 x N=512 (x read from HBM once).
// Wave (wm = wave/4 in 0..1, wn = wave%4 in 0..3): 32 M-rows x 128 N-cols
//   -> 2x8 accumulators of v8f, 16 v_wmma_f32_16x16x32_bf16 per K-chunk.
// ---------------------------------------------------------------------------
__global__ __launch_bounds__(256) void gemm_wmma_kernel(
    const float* __restrict__ x, const unsigned short* __restrict__ Psw,
    float* __restrict__ out)
{
  // A-panel in WMMA A-operand layout: index ((row*2 + half)*16 + e),
// where k = e + 16*half
  __shared__ __align__(32) unsigned short lds_a[GM * 32];        // 4 KB
  __shared__ __align__(32) unsigned short lds_b[32 * 32 * 16];   // 32 KB

  const int t    = threadIdx.x;
  const int wave = t >> 5;
  const int lane = t & 31;
  const int wm   = wave >> 2;            // 0..1
  const int wn   = wave & 3;             // 0..3
  const int bm   = blockIdx.x * GM;

  v8f acc[2][8];
  #pragma unroll
  for (int m = 0; m < 2; ++m)
    #pragma unroll
    for (int n = 0; n < 8; ++n)
      acc[m][n] = {};

  // A staging: each thread converts 8 consecutive f32 -> one 16B bf16 store.
  const int aidx = t * 8;                      // 0..2047 over the 64x32 panel
  const int arow = aidx >> 5;
  const int akb  = aidx & 31;                  // 0,8,16,24
  const int ah   = akb >> 4;
  const int ae   = akb & 15;
  unsigned short* asta = &lds_a[(((arow << 1) + ah) << 4) + ae];
  const float* xsrc = x + (size_t)(bm + arow) * NIN + akb;

  const int a_off0 = (((wm * 32 +  0 + (lane & 15)) << 1) + (lane >> 4)) << 4;
  const int a_off1 = (((wm * 32 + 16 + (lane & 15)) << 1) + (lane >> 4)) << 4;

  for (int kt = 0; kt < 16; ++kt) {
    // prefetch next x panel into caches (global_prefetch_b8)
    __builtin_prefetch(xsrc + (kt + 1) * 32, 0, 3);

    // stage A: 64x32 f32 -> bf16 (RNE), swizzled
    const float4 p0 = *(const float4*)(xsrc + kt * 32);
    const float4 p1 = *(const float4*)(xsrc + kt * 32 + 4);
    __align__(16) unsigned short tmp[8] = {
      f2bf(p0.x), f2bf(p0.y), f2bf(p0.z), f2bf(p0.w),
      f2bf(p1.x), f2bf(p1.y), f2bf(p1.z), f2bf(p1.w) };
    *(uint4*)asta = *(const uint4*)tmp;

    // stage B: straight 32 KB copy of the pre-swizzled P K-chunk (L2 resident)
    const uint4* bsrc = (const uint4*)(Psw + (size_t)kt * (32 * 32 * 16));
    uint4* bdst = (uint4*)lds_b;
    #pragma unroll
    for (int i = 0; i < 8; ++i) bdst[t + i * 256] = bsrc[t + i * 256];
    __syncthreads();

    const v16bf a0 = *(const v16bf*)&lds_a[a_off0];
    const v16bf a1 = *(const v16bf*)&lds_a[a_off1];
    #pragma unroll
    for (int n = 0; n < 8; ++n) {
      const v16bf b = *(const v16bf*)&lds_b[(((wn * 8 + n) * 32) + lane) << 4];
      acc[0][n] = __builtin_amdgcn_wmma_f32_16x16x32_bf16(
          false, a0, false, b, (short)0, acc[0][n], false, false);
      acc[1][n] = __builtin_amdgcn_wmma_f32_16x16x32_bf16(
          false, a1, false, b, (short)0, acc[1][n], false, false);
    }
    __syncthreads();
  }

  // Epilogue per C/D layout: VGPR r -> M = r + 8*(lane>>4), N = lane&15.
  const int colbase = wn * 128 + (lane & 15);
  const int rowbase = bm + wm * 32 + ((lane >> 4) << 3);
  #pragma unroll
  for (int m = 0; m < 2; ++m)
    #pragma unroll
    for (int n = 0; n < 8; ++n) {
      const int col = colbase + n * 16;
      #pragma unroll
      for (int r = 0; r < 8; ++r)
        out[(size_t)(rowbase + m * 16 + r) * NLV + col] = acc[m][n][r];
    }
}

// ---------------------------------------------------------------------------
extern "C" void kernel_launch(void* const* d_in, const int* in_sizes, int n_in,
                              void* d_out, int out_size, void* d_ws, size_t ws_size,
                              hipStream_t stream) {
  const float* x      = (const float*)d_in[0];
  const float* logits = (const float*)d_in[1];
  const float* noise  = (const float*)d_in[2];
  float* out = (float*)d_out;

  // Workspace layout:
  //   [0, 512KB)          : Psw  (bf16 P, WMMA-B swizzled, 512x512)
  //   [512KB, +40KB)      : colsum[ITERS][512]  (f32)
  //   next 20*4 bytes     : barct[ITERS]        (int)
  unsigned short* Psw = (unsigned short*)d_ws;
  float* colsum = (float*)((char*)d_ws + 512 * 1024);
  int*   barct  = (int*)((char*)d_ws + 512 * 1024 + ITERS * NIN * sizeof(float));

  init_ws_kernel<<<(ITERS * NIN + 255) / 256, 256, 0, stream>>>(colsum, barct);
  sinkhorn_kernel<<<SBLK, 512, 0, stream>>>(logits, noise, Psw, colsum, barct);
  gemm_wmma_kernel<<<BATCH / GM, 256, 0, stream>>>(x, Psw, out);
}
